// SinkhornRegularizer_26680336842874
// MI455X (gfx1250) — matrix operand order repaired and checked by reading, not compiled, not run on hardware
//
#include <hip/hip_runtime.h>
#include <hip/hip_bf16.h>
#include <math.h>

// ---------------------------------------------------------------------------
// Sinkhorn regularizer, MI455X (gfx1250, wave32).
//   C[i,j] = max(2 - 2 * dot(xn_i, yn_j), 0)   (unit vectors => sq terms = 1)
//   log_K  = -C / eps  (eps = 0.2) folded into the streaming sweeps.
//
// GEMM: bf16x3 split-precision WMMA (v_wmma_f32_16x16x32_bf16); operands are
// L2-resident (33.5 MB << 192 MB L2).
//
// Sweeps: 22 passes over the 256 MB cost matrix. L2 partitioning via temporal
// hints: rows [0, 5120) (160 MB) stay L2-resident (RT), rows [5120, 8192)
// (96 MB) are streamed non-temporally (TH=NT) so they never evict the
// resident slice. HBM traffic per sweep: 256 MB -> ~96 MB.
// ---------------------------------------------------------------------------

typedef __attribute__((ext_vector_type(16))) __bf16 v16bf;
typedef __attribute__((ext_vector_type(8)))  __bf16 v8bf;
typedef __attribute__((ext_vector_type(8)))  float  v8f;
typedef __attribute__((ext_vector_type(4)))  float  v4f;

#define NROWS 8192
#define DDIM  512
#define INV_EPS 5.0f          // 1 / (0.1 * 2)
#define SINK_ITERS 10
#define RES_ROWS 5120         // 160 MB resident slice (mult of 128 and 256)
#define COL_CHUNK 256         // rows per column-pass chunk
#define N_CHUNKS (NROWS / COL_CHUNK)          // 32
#define RES_CHUNKS (RES_ROWS / COL_CHUNK)     // 20

template <bool NT>
__device__ __forceinline__ v4f ld4(const v4f* p)
{
    if (NT) return __builtin_nontemporal_load(p);
    return *p;
}

__device__ __forceinline__ void lse_update(float& m, float& s, float e)
{
    float M = fmaxf(m, e);
    s = s * __expf(m - M) + __expf(e - M);
    m = M;
}

__device__ __forceinline__ void lse_merge(float& m, float& s, float m2, float s2)
{
    float M = fmaxf(m, m2);
    s = s * __expf(m - M) + s2 * __expf(m2 - M);
    m = M;
}

// ---------------------------- normalize + split ----------------------------
__global__ __launch_bounds__(256) void k_normalize_split(
    const float* __restrict__ X, const float* __restrict__ Y,
    __bf16* __restrict__ Ah, __bf16* __restrict__ Al,
    __bf16* __restrict__ Bh, __bf16* __restrict__ Bl)
{
    __shared__ float red[256];
    const int row = blockIdx.x;
    const int t   = threadIdx.x;
    const size_t base = (size_t)row * DDIM;

    float x0 = X[base + t], x1 = X[base + t + 256];
    float y0 = Y[base + t], y1 = Y[base + t + 256];

    red[t] = x0 * x0 + x1 * x1;
    __syncthreads();
    for (int s = 128; s > 0; s >>= 1) { if (t < s) red[t] += red[t + s]; __syncthreads(); }
    float sx = 1.0f / fmaxf(sqrtf(red[0]), 1e-12f);
    __syncthreads();
    red[t] = y0 * y0 + y1 * y1;
    __syncthreads();
    for (int s = 128; s > 0; s >>= 1) { if (t < s) red[t] += red[t + s]; __syncthreads(); }
    float sy = 1.0f / fmaxf(sqrtf(red[0]), 1e-12f);

    float xn0 = x0 * sx, xn1 = x1 * sx, yn0 = y0 * sy, yn1 = y1 * sy;
    __bf16 h;
    h = (__bf16)xn0; Ah[base + t]       = h; Al[base + t]       = (__bf16)(xn0 - (float)h);
    h = (__bf16)xn1; Ah[base + t + 256] = h; Al[base + t + 256] = (__bf16)(xn1 - (float)h);
    h = (__bf16)yn0; Bh[base + t]       = h; Bl[base + t]       = (__bf16)(yn0 - (float)h);
    h = (__bf16)yn1; Bh[base + t + 256] = h; Bl[base + t + 256] = (__bf16)(yn1 - (float)h);
}

// ------------------------------- WMMA GEMM ---------------------------------
// C = max(2 - 2*A.B^T, 0).  Block tile 128x128, 8 waves, wave tile 32x64.
__global__ __launch_bounds__(256) void k_gemm_cost(
    const __bf16* __restrict__ Ah, const __bf16* __restrict__ Al,
    const __bf16* __restrict__ Bh, const __bf16* __restrict__ Bl,
    float* __restrict__ C)
{
    const int lane = threadIdx.x & 31;
    const int wave = threadIdx.x >> 5;
    const int wm   = wave & 3;             // 0..3  -> M offset wm*32
    const int wn   = wave >> 2;            // 0..1  -> N offset wn*64
    const int l16  = lane & 15;
    const int hiL  = lane >> 4;            // 0 | 1
    const int ahk  = hiL << 4;             // A frag k base: lanes>=16 -> +16
    const int bhk  = hiL << 4;             // B frag k base: lanes>=16 -> +16

    const int blockM = blockIdx.y * 128;
    const int blockN = blockIdx.x * 128;

    v8f acc[2][4];
#pragma unroll
    for (int mt = 0; mt < 2; ++mt)
#pragma unroll
        for (int nt = 0; nt < 4; ++nt)
            acc[mt][nt] = (v8f)(0.0f);

    size_t aoff[2], boff[4];
#pragma unroll
    for (int mt = 0; mt < 2; ++mt)
        aoff[mt] = (size_t)(blockM + wm * 32 + mt * 16 + l16) * DDIM;
#pragma unroll
    for (int nt = 0; nt < 4; ++nt)
        boff[nt] = (size_t)(blockN + wn * 64 + nt * 16 + l16) * DDIM;

    for (int k0 = 0; k0 < DDIM; k0 += 32) {
        v16bf af_h[2], af_l[2], bf_h[4], bf_l[4];
        // A-matrix 16x32 bf16 layout: elems 0..7 -> k0+ahk+0..7,
        //                             elems 8..15 -> k0+ahk+8..15
#pragma unroll
        for (int mt = 0; mt < 2; ++mt) {
            const __bf16* pah = Ah + aoff[mt] + k0 + ahk;
            const __bf16* pal = Al + aoff[mt] + k0 + ahk;
            v8bf h0 = *(const v8bf*)(pah);
            v8bf h1 = *(const v8bf*)(pah + 8);
            v8bf l0 = *(const v8bf*)(pal);
            v8bf l1 = *(const v8bf*)(pal + 8);
            af_h[mt] = __builtin_shufflevector(h0, h1, 0,1,2,3,4,5,6,7,8,9,10,11,12,13,14,15);
            af_l[mt] = __builtin_shufflevector(l0, l1, 0,1,2,3,4,5,6,7,8,9,10,11,12,13,14,15);
        }
        // B-matrix 32x16 bf16 layout: lanes 0-15 k=0..15, lanes 16-31 k=16..31
#pragma unroll
        for (int nt = 0; nt < 4; ++nt) {
            bf_h[nt] = *(const v16bf*)(Bh + boff[nt] + k0 + bhk);
            bf_l[nt] = *(const v16bf*)(Bl + boff[nt] + k0 + bhk);
        }
        // bf16x3: hi*hi + hi*lo + lo*hi
#pragma unroll
        for (int mt = 0; mt < 2; ++mt)
#pragma unroll
            for (int nt = 0; nt < 4; ++nt) {
                acc[mt][nt] = __builtin_amdgcn_wmma_f32_16x16x32_bf16(
                    false, af_h[mt], false, bf_h[nt], (short)0, acc[mt][nt], false, false);
                acc[mt][nt] = __builtin_amdgcn_wmma_f32_16x16x32_bf16(
                    false, af_h[mt], false, bf_l[nt], (short)0, acc[mt][nt], false, false);
                acc[mt][nt] = __builtin_amdgcn_wmma_f32_16x16x32_bf16(
                    false, af_l[mt], false, bf_h[nt], (short)0, acc[mt][nt], false, false);
            }
    }

    // C/D layout: VGPR r -> M = r + (lane>=16 ? 8 : 0), N = lane&15
    // Resident rows stored RT (stay in L2); streaming rows stored NT.
    const bool resident = (blockM < RES_ROWS);
#pragma unroll
    for (int mt = 0; mt < 2; ++mt)
#pragma unroll
        for (int nt = 0; nt < 4; ++nt) {
            const int col = blockN + wn * 64 + nt * 16 + l16;
#pragma unroll
            for (int r = 0; r < 8; ++r) {
                const int row = blockM + wm * 32 + mt * 16 + r + (hiL << 3);
                float v = fmaxf(2.0f - 2.0f * acc[mt][nt][r], 0.0f);
                float* dst = C + (size_t)row * NROWS + col;
                if (resident) *dst = v;
                else          __builtin_nontemporal_store(v, dst);
            }
        }
}

// ------------------------------ init log_u/v -------------------------------
__global__ void k_init_logs(float* __restrict__ logu, float* __restrict__ logv)
{
    int i = blockIdx.x * blockDim.x + threadIdx.x;
    if (i < NROWS) { logu[i] = 0.0f; logv[i] = 0.0f; }
}

// --------------------- row LSE: log_u = -lse_j(lv_j - C/eps) ---------------
template <bool NT>
__global__ __launch_bounds__(256) void k_lse_rows(
    const float* __restrict__ C, const float* __restrict__ logv,
    float* __restrict__ logu, int row_base)
{
    const int i = row_base + blockIdx.x, t = threadIdx.x;
    const v4f* crow = (const v4f*)(C + (size_t)i * NROWS);
    const v4f* lv4  = (const v4f*)logv;

    float m = -__builtin_inff(), s = 0.0f;
    for (int j4 = t; j4 < NROWS / 4; j4 += 256) {
        v4f c = ld4<NT>(crow + j4);
        v4f v = lv4[j4];
        lse_update(m, s, v.x - c.x * INV_EPS);
        lse_update(m, s, v.y - c.y * INV_EPS);
        lse_update(m, s, v.z - c.z * INV_EPS);
        lse_update(m, s, v.w - c.w * INV_EPS);
    }
    __shared__ float sm[256], ss[256];
    sm[t] = m; ss[t] = s; __syncthreads();
    for (int w = 128; w > 0; w >>= 1) {
        if (t < w) {
            float mm = sm[t], sv = ss[t];
            lse_merge(mm, sv, sm[t + w], ss[t + w]);
            sm[t] = mm; ss[t] = sv;
        }
        __syncthreads();
    }
    if (t == 0) logu[i] = -(sm[0] + __logf(ss[0]));
}

// ---------------- column LSE partials (float4-coalesced) -------------------
// Block: 256 threads x 4 columns = 1024 columns, one 256-row chunk.
template <bool NT>
__global__ __launch_bounds__(256) void k_lse_cols_partial(
    const float* __restrict__ C, const float* __restrict__ logu,
    float* __restrict__ pm, float* __restrict__ ps, int chunk_base)
{
    const int c4    = blockIdx.x * 256 + threadIdx.x;   // v4f column index
    const int chunk = chunk_base + blockIdx.y;
    const int r0    = chunk * COL_CHUNK;
    const v4f* C4 = (const v4f*)C;

    float m0 = -__builtin_inff(), m1 = m0, m2 = m0, m3 = m0;
    float s0 = 0.f, s1 = 0.f, s2 = 0.f, s3 = 0.f;
#pragma unroll 4
    for (int r = r0; r < r0 + COL_CHUNK; ++r) {
        float lu = logu[r];
        v4f   c  = ld4<NT>(C4 + (size_t)r * (NROWS / 4) + c4);
        lse_update(m0, s0, lu - c.x * INV_EPS);
        lse_update(m1, s1, lu - c.y * INV_EPS);
        lse_update(m2, s2, lu - c.z * INV_EPS);
        lse_update(m3, s3, lu - c.w * INV_EPS);
    }
    const size_t o = (size_t)chunk * (NROWS / 4) + c4;
    v4f vm; vm.x = m0; vm.y = m1; vm.z = m2; vm.w = m3;
    v4f vs; vs.x = s0; vs.y = s1; vs.z = s2; vs.w = s3;
    ((v4f*)pm)[o] = vm;
    ((v4f*)ps)[o] = vs;
}

__global__ __launch_bounds__(256) void k_lse_cols_combine(
    const float* __restrict__ pm, const float* __restrict__ ps,
    float* __restrict__ logv)
{
    const int j = blockIdx.x * 256 + threadIdx.x;
    float m = -__builtin_inff(), s = 0.0f;
#pragma unroll
    for (int p = 0; p < N_CHUNKS; ++p)
        lse_merge(m, s, pm[(size_t)p * NROWS + j], ps[(size_t)p * NROWS + j]);
    logv[j] = -(m + __logf(s));
}

// ----------------- final: rowsum_i = sum_j clamp(P)*C ----------------------
template <bool NT>
__global__ __launch_bounds__(256) void k_psum_rows(
    const float* __restrict__ C, const float* __restrict__ logu,
    const float* __restrict__ logv, float* __restrict__ rowsum, int row_base)
{
    const int i = row_base + blockIdx.x, t = threadIdx.x;
    const float lu = logu[i];
    const v4f* crow = (const v4f*)(C + (size_t)i * NROWS);
    const v4f* lv4  = (const v4f*)logv;

    float acc = 0.0f;
    for (int j4 = t; j4 < NROWS / 4; j4 += 256) {
        v4f c = ld4<NT>(crow + j4);
        v4f v = lv4[j4];
        acc += fminf(__expf(lu + v.x - c.x * INV_EPS), 1.0f) * c.x;
        acc += fminf(__expf(lu + v.y - c.y * INV_EPS), 1.0f) * c.y;
        acc += fminf(__expf(lu + v.z - c.z * INV_EPS), 1.0f) * c.z;
        acc += fminf(__expf(lu + v.w - c.w * INV_EPS), 1.0f) * c.w;
    }
    __shared__ float red[256];
    red[t] = acc; __syncthreads();
    for (int s = 128; s > 0; s >>= 1) { if (t < s) red[t] += red[t + s]; __syncthreads(); }
    if (t == 0) rowsum[i] = red[0];
}

__global__ __launch_bounds__(256) void k_finalize(
    const float* __restrict__ C, const float* __restrict__ rowsum,
    float* __restrict__ out)
{
    const int t = threadIdx.x;
    float a = 0.0f, d = 0.0f;
    for (int i = t; i < NROWS; i += 256) {
        a += rowsum[i];
        d += C[(size_t)i * NROWS + i];     // diag: C_ii = 2 - 2*cos_ii
    }
    __shared__ float ra[256], rd[256];
    ra[t] = a; rd[t] = d; __syncthreads();
    for (int s = 128; s > 0; s >>= 1) {
        if (t < s) { ra[t] += ra[t + s]; rd[t] += rd[t + s]; }
        __syncthreads();
    }
    if (t == 0) {
        float dist = ra[0] / (float)NROWS;
        float fb   = (rd[0] / (float)NROWS) * 0.5f;   // 1 - mean(cos) = mean(C_ii)/2
        out[0] = (isnan(dist) || isinf(dist)) ? fb : dist;
    }
}

// ------------------------------- launch ------------------------------------
extern "C" void kernel_launch(void* const* d_in, const int* in_sizes, int n_in,
                              void* d_out, int out_size, void* d_ws, size_t ws_size,
                              hipStream_t stream)
{
    const float* X = (const float*)d_in[0];
    const float* Y = (const float*)d_in[1];

    char* ws = (char*)d_ws;
    const size_t SZ_C  = (size_t)NROWS * NROWS * sizeof(float);   // 256 MiB
    const size_t SZ_BF = (size_t)NROWS * DDIM * sizeof(__bf16);   // 8 MiB each
    float*  C    = (float*)(ws);
    __bf16* Ah   = (__bf16*)(ws + SZ_C);
    __bf16* Al   = (__bf16*)(ws + SZ_C + SZ_BF);
    __bf16* Bh   = (__bf16*)(ws + SZ_C + 2 * SZ_BF);
    __bf16* Bl   = (__bf16*)(ws + SZ_C + 3 * SZ_BF);
    char*   p    = ws + SZ_C + 4 * SZ_BF;
    float*  logu = (float*)(p);                 p += NROWS * sizeof(float);
    float*  logv = (float*)(p);                 p += NROWS * sizeof(float);
    float*  pm   = (float*)(p);                 p += (size_t)N_CHUNKS * NROWS * sizeof(float);
    float*  ps   = (float*)(p);                 p += (size_t)N_CHUNKS * NROWS * sizeof(float);
    float*  rsum = (float*)(p);

    k_normalize_split<<<NROWS, 256, 0, stream>>>(X, Y, Ah, Al, Bh, Bl);
    k_gemm_cost<<<dim3(64, 64), 256, 0, stream>>>(Ah, Al, Bh, Bl, C);
    k_init_logs<<<(NROWS + 255) / 256, 256, 0, stream>>>(logu, logv);

    for (int it = 0; it < SINK_ITERS; ++it) {
        // log_u update: resident rows (RT) + streaming rows (NT)
        k_lse_rows<false><<<RES_ROWS, 256, 0, stream>>>(C, logv, logu, 0);
        k_lse_rows<true ><<<NROWS - RES_ROWS, 256, 0, stream>>>(C, logv, logu, RES_ROWS);
        // log_v update: column partials over 256-row chunks, then combine
        k_lse_cols_partial<false><<<dim3(8, RES_CHUNKS), 256, 0, stream>>>(
            C, logu, pm, ps, 0);
        k_lse_cols_partial<true ><<<dim3(8, N_CHUNKS - RES_CHUNKS), 256, 0, stream>>>(
            C, logu, pm, ps, RES_CHUNKS);
        k_lse_cols_combine<<<NROWS / 256, 256, 0, stream>>>(pm, ps, logv);
    }

    k_psum_rows<false><<<RES_ROWS, 256, 0, stream>>>(C, logu, logv, rsum, 0);
    k_psum_rows<true ><<<NROWS - RES_ROWS, 256, 0, stream>>>(C, logu, logv, rsum, RES_ROWS);
    k_finalize<<<1, 256, 0, stream>>>(C, rsum, (float*)d_out);
}
